// LSTM_attention_71554155151447
// MI455X (gfx1250) — compile-verified, matrix operand order checked by reading, not run on hardware
//
#include <hip/hip_runtime.h>
#include <hip/hip_bf16.h>
#include <math.h>

// Problem constants (match reference)
#define V1N  32000
#define V2N  32000
#define HIDN 128
#define G4   512     // 4*HID
#define T1N  2048
#define T2N  2048

typedef __attribute__((ext_vector_type(16))) __bf16 v16bf;
typedef __attribute__((ext_vector_type(8)))  __bf16 v8bf;
typedef __attribute__((ext_vector_type(4)))  __bf16 v4bf;
typedef __attribute__((ext_vector_type(8)))  float  v8f;

// ISA 7.12.2 16-bit A/B fragment K-index for lane `lane`, vector element `e`:
// K = (lane/16)*8 + (e/8)*16 + (e%8)  -> two contiguous 8-element runs per lane
__device__ __forceinline__ int wk(int lane, int e) {
  return ((lane >> 4) << 3) + ((e >> 3) << 4) + (e & 7);
}

// Build a 16x32 bf16 fragment from two contiguous 8-element (16B) runs.
__device__ __forceinline__ v16bf frag_from(const __bf16* __restrict__ p) {
  const v8bf lo = *(const v8bf*)p;        // K = hi8 + 0..7
  const v8bf hi = *(const v8bf*)(p + 16); // K = hi8 + 16..23
  return __builtin_shufflevector(lo, hi, 0, 1, 2, 3, 4, 5, 6, 7,
                                         8, 9, 10, 11, 12, 13, 14, 15);
}

__device__ __forceinline__ float sigf(float x) { return 1.0f / (1.0f + __expf(-x)); }

// ---------------------------------------------------------------------------
// Kernel 1: gather embedding columns  out[t][j] = M[j][idx[t]]
// ---------------------------------------------------------------------------
__global__ void gather_cols_kernel(const float* __restrict__ M,
                                   const int* __restrict__ idx,
                                   float* __restrict__ out, int vocab) {
  const int t = blockIdx.x;
  const int j = threadIdx.x;                 // 512 threads
  out[(size_t)t * G4 + j] = M[(size_t)j * vocab + idx[t]];
}

// ---------------------------------------------------------------------------
// Kernel 2: f32 -> bf16 bulk convert (for V), 4 elements/thread
// ---------------------------------------------------------------------------
__global__ void cvt_bf16_kernel(const float* __restrict__ src,
                                __bf16* __restrict__ dst, int n4) {
  const int i = blockIdx.x * blockDim.x + threadIdx.x;
  if (i < n4) {
    const float4 v = ((const float4*)src)[i];
    v4bf o = { (__bf16)v.x, (__bf16)v.y, (__bf16)v.z, (__bf16)v.w };
    ((v4bf*)dst)[i] = o;
  }
}

// ---------------------------------------------------------------------------
// Kernel 3: both LSTM scans, single persistent workgroup (512 thr = 16 waves).
// Weight matrix transposed into 256 KB LDS (CDNA5: 320KB per WGP).
// ---------------------------------------------------------------------------
__global__ void lstm_kernel(const float* __restrict__ W, const float* __restrict__ W_,
                            const float* __restrict__ Ux, const float* __restrict__ Uy,
                            float* __restrict__ hs, float* __restrict__ ss) {
  extern __shared__ float Wl[];              // [128][512] transposed, 256 KB
  __shared__ float hsh[HIDN];
  __shared__ float csh[HIDN];
  __shared__ float gates[G4];
  const int tid = threadIdx.x;               // 0..511

  if (tid < HIDN) { hsh[tid] = 0.0f; csh[tid] = 0.0f; }

  for (int phase = 0; phase < 2; ++phase) {
    const float* Wsrc = phase ? W_ : W;
    const float* Uin  = phase ? Uy : Ux;
    float*       hout = phase ? ss : hs;
    const int    T    = phase ? T2N : T1N;

    __syncthreads();
    for (int i = tid; i < G4 * HIDN; i += G4) {   // Wl[k*512+j] = Wsrc[j*128+k]
      const int j = i >> 7, k = i & 127;
      Wl[k * G4 + j] = Wsrc[i];
    }
    __syncthreads();

    for (int t = 0; t < T; ++t) {
      float acc = Uin[(size_t)t * G4 + tid];
#pragma unroll 8
      for (int k = 0; k < HIDN; ++k) acc = fmaf(Wl[k * G4 + tid], hsh[k], acc);
      gates[tid] = acc;
      __syncthreads();
      if (tid < HIDN) {
        const float ig = sigf(gates[tid]);
        const float fg = sigf(gates[HIDN + tid]);
        const float og = sigf(gates[2 * HIDN + tid]);
        const float gg = tanhf(gates[3 * HIDN + tid]);
        const float c  = fg * csh[tid] + ig * gg;
        const float h  = og * tanhf(c);
        csh[tid] = c; hsh[tid] = h;
        hout[(size_t)t * HIDN + tid] = h;
      }
      __syncthreads();
    }
  }
}

// ---------------------------------------------------------------------------
// Kernel 4: pack encoder states: hsb = bf16(hs) row-major, hsT = bf16(hs)^T
// ---------------------------------------------------------------------------
__global__ void hs_pack_kernel(const float* __restrict__ hs,
                               __bf16* __restrict__ hsb,
                               __bf16* __restrict__ hsT) {
  const int t = blockIdx.x;                  // 0..2047
  const int h = threadIdx.x;                 // 0..127
  const float v = hs[(size_t)t * HIDN + h];
  hsb[(size_t)t * HIDN + h] = (__bf16)v;
  hsT[(size_t)h * T1N + t]  = (__bf16)v;
}

// ---------------------------------------------------------------------------
// Kernel 5: attention for a block of 16 decoder rows.
//   E = ss @ hs^T  (WMMA bf16, scores kept in 128KB LDS)
//   softmax rows in LDS
//   A = alpha @ hs (WMMA bf16 against transposed bf16 hs)
//   emit result = [A | ss] as bf16 into workspace
// ---------------------------------------------------------------------------
__global__ void attention_kernel(const __bf16* __restrict__ hsb,
                                 const __bf16* __restrict__ hsT,
                                 const float* __restrict__ ss,
                                 __bf16* __restrict__ resbf) {
  extern __shared__ float Esh[];             // [16][2048] f32 scores (128 KB)
  __shared__ float ssl[16 * HIDN];
  __shared__ float red[16][8];
  __shared__ float rowmax[16];
  __shared__ float rowsum[16];

  const int tid  = threadIdx.x;              // 128 threads = 4 waves
  const int lane = tid & 31;
  const int wave = tid >> 5;
  const int l16  = lane & 15;
  const int hi8  = (lane >> 4) << 3;
  const int m0   = blockIdx.x * 16;

  for (int i = tid; i < 16 * HIDN; i += 128) ssl[i] = ss[(size_t)m0 * HIDN + i];
  __syncthreads();

  // A-fragments of ss rows, K = HID split into 4 chunks of 32 (from LDS f32)
  v16bf assf[4];
#pragma unroll
  for (int kc = 0; kc < 4; ++kc) {
#pragma unroll
    for (int e = 0; e < 16; ++e)
      assf[kc][e] = (__bf16)ssl[l16 * HIDN + kc * 32 + wk(lane, e)];
  }

  // ---- Phase 1: scores E over this wave's 512 encoder columns ----
  const int col0 = wave * (T1N / 4);
  for (int tt = 0; tt < (T1N / 4) / 16; ++tt) {
    const int tb = col0 + tt * 16;
    v8f acc = {};
#pragma unroll
    for (int kc = 0; kc < 4; ++kc) {
      // B(k,n) = hs[tb+n][kc*32+k]: two 16B loads per lane
      const v16bf b = frag_from(&hsb[(size_t)(tb + l16) * HIDN + kc * 32 + hi8]);
      acc = __builtin_amdgcn_wmma_f32_16x16x32_bf16(false, assf[kc], false, b,
                                                    (short)0, acc, false, false);
    }
#pragma unroll
    for (int r = 0; r < 8; ++r)
      Esh[(hi8 + r) * T1N + tb + l16] = acc[r];
  }
  __syncthreads();

  // ---- Phase 2: row softmax over T1 (8 threads per row) ----
  {
    const int r = tid >> 3, sub = tid & 7;
    float m = -1e30f;
    for (int n = sub; n < T1N; n += 8) m = fmaxf(m, Esh[r * T1N + n]);
    red[r][sub] = m;
    __syncthreads();
    if (sub == 0) {
      float mm = red[r][0];
      for (int q = 1; q < 8; ++q) mm = fmaxf(mm, red[r][q]);
      rowmax[r] = mm;
    }
    __syncthreads();
    const float mm = rowmax[r];
    float s = 0.0f;
    for (int n = sub; n < T1N; n += 8) s += __expf(Esh[r * T1N + n] - mm);
    red[r][sub] = s;
    __syncthreads();
    if (sub == 0) {
      float t = 0.0f;
      for (int q = 0; q < 8; ++q) t += red[r][q];
      rowsum[r] = t;
    }
    __syncthreads();
    const float inv = 1.0f / rowsum[r];
    for (int n = sub; n < T1N; n += 8)
      Esh[r * T1N + n] = __expf(Esh[r * T1N + n] - mm) * inv;
  }
  __syncthreads();

  // ---- Phase 3: A = alpha @ hs, per-wave partial over its K-range ----
  v8f acc8[8] = {};
  for (int cc = 0; cc < (T1N / 4) / 32; ++cc) {
    const int cb = col0 + cc * 32;
    v16bf a;                                 // A(m,k) = alpha[m][cb+k] from LDS
#pragma unroll
    for (int e = 0; e < 16; ++e)
      a[e] = (__bf16)Esh[l16 * T1N + cb + wk(lane, e)];
#pragma unroll
    for (int ht = 0; ht < 8; ++ht) {
      // B(k,n) = hs[cb+k][ht*16+n] = hsT[ht*16+n][cb+k]: two 16B loads
      const v16bf b = frag_from(&hsT[(size_t)(ht * 16 + l16) * T1N + cb + hi8]);
      acc8[ht] = __builtin_amdgcn_wmma_f32_16x16x32_bf16(false, a, false, b,
                                                         (short)0, acc8[ht], false, false);
    }
  }
  __syncthreads();                           // done reading Esh; reuse it
#pragma unroll
  for (int ht = 0; ht < 8; ++ht) {
#pragma unroll
    for (int r = 0; r < 8; ++r)
      Esh[wave * 2048 + (hi8 + r) * HIDN + ht * 16 + l16] = acc8[ht][r];
  }
  __syncthreads();
  for (int i = tid; i < 2048; i += 128) {
    const float sum = Esh[i] + Esh[2048 + i] + Esh[4096 + i] + Esh[6144 + i];
    const int m = i >> 7, h = i & 127;
    resbf[(size_t)(m0 + m) * 256 + h]       = (__bf16)sum;               // context A
    resbf[(size_t)(m0 + m) * 256 + 128 + h] = (__bf16)ssl[m * HIDN + h]; // ss
  }
}

// ---------------------------------------------------------------------------
// Kernel 6: logits = result @ V^T (bf16 WMMA, f32 out). 16x64 tile per wave.
// A rows (16x256 bf16 = 8KB) staged to LDS via async-to-LDS DMA path.
// ---------------------------------------------------------------------------
__global__ void logits_kernel(const __bf16* __restrict__ resbf,
                              const __bf16* __restrict__ Vbf,
                              float* __restrict__ out) {
  __shared__ __bf16 Abuf[16 * 256];          // only LDS object => offset 0
  const int tid  = threadIdx.x;              // 128 threads = 4 waves
  const int lane = tid & 31;
  const int wave = tid >> 5;
  const int l16  = lane & 15;
  const int hi8  = (lane >> 4) << 3;
  const int m0   = blockIdx.y * 16;
  const int n0   = blockIdx.x * 256 + wave * 64;

  // Async stage of the 8KB A-tile into LDS: 128 threads x 4 x 16B (ASYNCcnt path)
  {
    const unsigned long long gbase = (unsigned long long)(resbf + (size_t)m0 * 256);
    const unsigned off0 = (unsigned)tid * 16u;     // byte offset (== LDS offset)
#pragma unroll
    for (int i = 0; i < 4; ++i) {
      const unsigned off = off0 + (unsigned)i * 2048u;
      asm volatile("global_load_async_to_lds_b128 %0, %1, %2"
                   :: "v"(off), "v"(off), "s"(gbase) : "memory");
    }
    asm volatile("s_wait_asynccnt 0x0" ::: "memory");
  }
  __syncthreads();

  v16bf a[8];                                // K = 256 in 8 chunks of 32
#pragma unroll
  for (int kc = 0; kc < 8; ++kc)
    a[kc] = frag_from(&Abuf[l16 * 256 + kc * 32 + hi8]);   // two ds_load_b128

  for (int nt = 0; nt < 4; ++nt) {
    const int nc = n0 + nt * 16;
    if (nt < 3)                              // CDNA5 global_prefetch of next V tile
      __builtin_prefetch(&Vbf[(size_t)(nc + 16 + l16) * 256], 0, 1);
    v8f acc = {};
#pragma unroll
    for (int kc = 0; kc < 8; ++kc) {
      // B(k,n) = V[nc+n][kc*32+k]: two global_load_b128 per lane
      const v16bf b = frag_from(&Vbf[(size_t)(nc + l16) * 256 + kc * 32 + hi8]);
      acc = __builtin_amdgcn_wmma_f32_16x16x32_bf16(false, a[kc], false, b,
                                                    (short)0, acc, false, false);
    }
#pragma unroll
    for (int r = 0; r < 8; ++r)
      out[(size_t)(m0 + hi8 + r) * V2N + nc + l16] = acc[r];   // coalesced
  }
}

// ---------------------------------------------------------------------------
// Kernel 7: per-row softmax over vocab (32000), in-place on d_out.
// ---------------------------------------------------------------------------
__global__ void softmax_rows_kernel(float* __restrict__ out) {
  __shared__ float red[256];
  const int tid = threadIdx.x;
  float* p = out + (size_t)blockIdx.x * V2N;

  float m = -1e30f;
  for (int i = tid; i < V2N; i += 256) m = fmaxf(m, p[i]);
  red[tid] = m; __syncthreads();
  for (int s = 128; s > 0; s >>= 1) {
    if (tid < s) red[tid] = fmaxf(red[tid], red[tid + s]);
    __syncthreads();
  }
  m = red[0];
  __syncthreads();

  float sum = 0.0f;
  for (int i = tid; i < V2N; i += 256) sum += __expf(p[i] - m);
  red[tid] = sum; __syncthreads();
  for (int s = 128; s > 0; s >>= 1) {
    if (tid < s) red[tid] += red[tid + s];
    __syncthreads();
  }
  const float inv = 1.0f / red[0];
  for (int i = tid; i < V2N; i += 256) p[i] = __expf(p[i] - m) * inv;
}

// ---------------------------------------------------------------------------
extern "C" void kernel_launch(void* const* d_in, const int* in_sizes, int n_in,
                              void* d_out, int out_size, void* d_ws, size_t ws_size,
                              hipStream_t stream) {
  const int*   x  = (const int*)d_in[0];
  const int*   y  = (const int*)d_in[1];
  const float* U  = (const float*)d_in[2];
  const float* W  = (const float*)d_in[3];
  const float* U_ = (const float*)d_in[4];
  const float* W_ = (const float*)d_in[5];
  const float* V  = (const float*)d_in[6];
  float* out = (float*)d_out;

  // Workspace layout (~28.5 MB)
  char* wsB = (char*)d_ws;
  float*  Ux    = (float*)wsB;   wsB += (size_t)T1N * G4 * sizeof(float);
  float*  Uy    = (float*)wsB;   wsB += (size_t)T2N * G4 * sizeof(float);
  float*  hs    = (float*)wsB;   wsB += (size_t)T1N * HIDN * sizeof(float);
  float*  ss    = (float*)wsB;   wsB += (size_t)T2N * HIDN * sizeof(float);
  __bf16* resbf = (__bf16*)wsB;  wsB += (size_t)T2N * 256 * sizeof(__bf16);
  __bf16* hsb   = (__bf16*)wsB;  wsB += (size_t)T1N * HIDN * sizeof(__bf16);
  __bf16* hsT   = (__bf16*)wsB;  wsB += (size_t)HIDN * T1N * sizeof(__bf16);
  __bf16* Vbf   = (__bf16*)wsB;  // V2N * 256 bf16 (16 MB)

  cvt_bf16_kernel<<<(V2N * 256 / 4 + 255) / 256, 256, 0, stream>>>(V, Vbf, V2N * 256 / 4);
  gather_cols_kernel<<<T1N, G4, 0, stream>>>(U,  x, Ux, V1N);
  gather_cols_kernel<<<T2N, G4, 0, stream>>>(U_, y, Uy, V2N);
  lstm_kernel<<<1, G4, G4 * HIDN * sizeof(float), stream>>>(W, W_, Ux, Uy, hs, ss);
  hs_pack_kernel<<<T1N, HIDN, 0, stream>>>(hs, hsb, hsT);
  attention_kernel<<<T2N / 16, 128, 16 * T1N * sizeof(float), stream>>>(hsb, hsT, ss, resbf);
  logits_kernel<<<dim3(V2N / 256, T2N / 16), 128, 0, stream>>>(resbf, Vbf, out);
  softmax_rows_kernel<<<T2N, 256, 0, stream>>>(out);
}
